// CorefTransformerLayer_26388279067454
// MI455X (gfx1250) — matrix operand order repaired and checked by reading, not compiled, not run on hardware
//
#include <hip/hip_runtime.h>

// ---------------- problem constants ----------------
#define BB   16
#define SS   512
#define DD   512
#define HH   8
#define DH   64
#define TCC  128
#define LLN  12
#define DCC  768
#define DFF2 2048
#define BS   (BB * SS)        // 8192
#define TCS  (TCC * SS)       // 65536
#define TCL  (TCC * LLN)      // 1536

typedef __bf16 bf16;
typedef __attribute__((ext_vector_type(16))) __bf16 v16bf;
typedef __attribute__((ext_vector_type(8)))  __bf16 v8bf;
typedef __attribute__((ext_vector_type(8)))  float  v8f;
typedef unsigned int v4u __attribute__((ext_vector_type(4)));
typedef int          v8i __attribute__((ext_vector_type(8)));
typedef int          v4i __attribute__((ext_vector_type(4)));

// LDS B-tile: 64 rows x 64 bf16, padded 16B per 128B row -> 72-element pitch
#define KBLK    64
#define BPITCH  72
#define BTILE_ELEMS (64 * BPITCH)

// ---------------- WMMA fragment helpers (CDNA5 wave32 layouts) ----------------
// 16-bit A 16x32 (MxK), this file's contiguous K order (not the ISA table): lanes 0-15 -> row=lane, K=[0..15];
//                       lanes 16-31 -> row=lane-16, K=[16..31].
__device__ __forceinline__ v16bf load_a_frag(const bf16* base, int ld, int lane) {
  int r = lane & 15, h = lane >> 4;
  const bf16* p = base + (size_t)r * ld + h * 16;
  v8bf lo = *(const v8bf*)(p);
  v8bf hi = *(const v8bf*)(p + 8);
  v16bf f;
#pragma unroll
  for (int i = 0; i < 8; ++i) { f[i] = lo[i]; f[i + 8] = hi[i]; }
  return f;
}
// 16-bit B 32x16 (KxN) from B^T stored row-major [N,K]:
// same contiguous K order as A: lanes 0-15 -> col=lane, K=[0..15]; lanes 16-31 -> col=lane-16, K=[16..31].
__device__ __forceinline__ v16bf load_b_frag(const bf16* baseT, int ld, int lane) {
  int n = lane & 15, h = lane >> 4;
  const bf16* p = baseT + (size_t)n * ld + h * 16;
  v8bf lo = *(const v8bf*)(p);
  v8bf hi = *(const v8bf*)(p + 8);
  v16bf f;
#pragma unroll
  for (int i = 0; i < 8; ++i) { f[i] = lo[i]; f[i + 8] = hi[i]; }
  return f;
}
// B fragment out of the padded LDS tile (rows = columns of B, pitch BPITCH)
__device__ __forceinline__ v16bf load_b_frag_lds(const bf16* buf, int j, int ks, int lane) {
  int n = lane & 15, h = lane >> 4;
  const bf16* p = buf + (size_t)(j * 16 + n) * BPITCH + ks + h * 16;
  v8bf lo = *(const v8bf*)(p);
  v8bf hi = *(const v8bf*)(p + 8);
  v16bf f;
#pragma unroll
  for (int i = 0; i < 8; ++i) { f[i] = lo[i]; f[i + 8] = hi[i]; }
  return f;
}
__device__ __forceinline__ v8f wmma_bf16(v16bf a, v16bf b, v8f c) {
  return __builtin_amdgcn_wmma_f32_16x16x32_bf16(false, a, false, b, (short)0, c,
                                                 false, false);
}
// C/D f32 16x16: lane holds col=lane&15; vgpr v holds row v + 8*(lane>>4).

// ---------------- Tensor Data Mover: 2D B-tile (64 rows x 64 bf16) -> LDS ----------------
#if __has_builtin(__builtin_amdgcn_tensor_load_to_lds) && \
    __has_builtin(__builtin_amdgcn_s_wait_tensorcnt)
#define USE_TDM 1
__device__ __forceinline__ void tdm_issue_b_tile(const bf16* gsrc, int ld_elems,
                                                 unsigned lds_byte_addr) {
  unsigned long long ga = (unsigned long long)(uintptr_t)gsrc;
  // D# group0: count=1 | lds_addr | global_addr[56:0] | type=2 ("image")
  v4u g0 = { 1u, lds_byte_addr, (unsigned)ga,
             (unsigned)((ga >> 32) & 0x01FFFFFFull) | 0x80000000u };
  // D# group1: data_size=2B, pad_enable, pad_interval=32dw(128B), pad_amount=4dw(16B)
  unsigned flags = (1u << 16) | (1u << 20) | (4u << 22) | (3u << 25);
  unsigned td0 = 0x7FFFFFFFu, td1 = 0x7FFFFFFFu;        // tensor dims: no clamping
  unsigned long long s0 = (unsigned long long)ld_elems; // dim0 stride (elements)
  v8i g1 = { (int)flags,
             (int)((td0 & 0xFFFFu) << 16),                         // td0[15:0]
             (int)((td0 >> 16) | ((td1 & 0xFFFFu) << 16)),         // td0[31:16], td1[15:0]
             (int)((td1 >> 16) | (64u << 16)),                     // td1[31:16], tile_dim0=64
             (int)64u,                                             // tile_dim1=64, tile_dim2=0
             (int)(unsigned)(s0 & 0xFFFFFFFFull),                  // stride0 lo
             (int)(unsigned)((s0 >> 32) & 0xFFFFull),              // stride0 hi, stride1 lo=0
             0 };
  v4i z4 = { 0, 0, 0, 0 };
#if defined(__clang_major__) && __clang_major__ >= 23
  v8i z8 = { 0, 0, 0, 0, 0, 0, 0, 0 };
  __builtin_amdgcn_tensor_load_to_lds(g0, g1, z4, z4, z8, 0);
#else
  __builtin_amdgcn_tensor_load_to_lds(g0, g1, z4, z4, 0);
#endif
}
#else
#define USE_TDM 0
#endif

// order-preserving float <-> u32 encoding for atomic segment-max
__device__ __forceinline__ unsigned enc_f32(float x) {
  unsigned b = __float_as_uint(x);
  return (b & 0x80000000u) ? ~b : (b | 0x80000000u);
}
__device__ __forceinline__ float dec_f32(unsigned e) {
  unsigned b = (e & 0x80000000u) ? (e & 0x7FFFFFFFu) : ~e;
  return __uint_as_float(b);
}

// ---------------- generic GEMM: C[M,N] = A[M,K](bf16) * Bt[N,K]^T + bias ----------------
// grid = (M/256, N/64), block = 256 (8 waves); wave -> 32x64 strip (8 accumulators).
// B strip staged per-block into LDS by the Tensor Data Mover, double-buffered over
// 64-wide K blocks; TENSORcnt + workgroup barrier orders producer/consumers.
__global__ void k_gemm(const bf16* __restrict__ A, int lda,
                       const bf16* __restrict__ Bt, int ldb,
                       const float* __restrict__ bias,
                       float* __restrict__ Cf, bf16* __restrict__ Cb, int ldc,
                       const float* __restrict__ resid, int K, int relu) {
  __shared__ __align__(16) bf16 ldsB[2][BTILE_ELEMS];
  int lane = threadIdx.x & 31, wave = threadIdx.x >> 5;
  int m0 = (blockIdx.x * 8 + wave) * 32;
  int n0 = blockIdx.y * 64;
  v8f acc[2][4];
#pragma unroll
  for (int mf = 0; mf < 2; ++mf)
#pragma unroll
    for (int j = 0; j < 4; ++j) acc[mf][j] = (v8f)(0.0f);

  const bf16* btile = Bt + (size_t)n0 * ldb;   // rows n0..n0+63 of B^T
  int nkb = K / KBLK;

#if USE_TDM
  if (threadIdx.x == 0)
    tdm_issue_b_tile(btile, ldb, (unsigned)(uintptr_t)&ldsB[0][0]);
#endif
  for (int kb = 0; kb < nkb; ++kb) {
#if USE_TDM
    if (threadIdx.x == 0) __builtin_amdgcn_s_wait_tensorcnt((short)0);
    __syncthreads();     // publish buffer kb; prior reads of other buffer done
    if (threadIdx.x == 0 && kb + 1 < nkb)
      tdm_issue_b_tile(btile + (kb + 1) * KBLK, ldb,
                       (unsigned)(uintptr_t)&ldsB[(kb + 1) & 1][0]);
#else
    // cooperative fallback: 256 threads copy 64x64 bf16 into padded LDS tile
    {
      __syncthreads();
      int r = threadIdx.x >> 2, c = (threadIdx.x & 3) * 16;
      const v8bf* src = (const v8bf*)(btile + (size_t)r * ldb + kb * KBLK + c);
      v8bf* dst = (v8bf*)(&ldsB[kb & 1][(size_t)r * BPITCH + c]);
      dst[0] = src[0];
      dst[1] = src[1];
      __syncthreads();
    }
#endif
    const bf16* bb = &ldsB[kb & 1][0];
    const bf16* arow0 = A + (size_t)m0 * lda + kb * KBLK;
    const bf16* arow1 = arow0 + (size_t)16 * lda;
#pragma unroll
    for (int ks = 0; ks < KBLK; ks += 32) {
      v16bf a0 = load_a_frag(arow0 + ks, lda, lane);
      v16bf a1 = load_a_frag(arow1 + ks, lda, lane);
#pragma unroll
      for (int j = 0; j < 4; ++j) {
        v16bf b = load_b_frag_lds(bb, j, ks, lane);
        acc[0][j] = wmma_bf16(a0, b, acc[0][j]);
        acc[1][j] = wmma_bf16(a1, b, acc[1][j]);
      }
    }
  }
  int half = lane >> 4, col = lane & 15;
#pragma unroll
  for (int mf = 0; mf < 2; ++mf)
#pragma unroll
    for (int j = 0; j < 4; ++j) {
      int n = n0 + j * 16 + col;
      float bv = bias ? bias[n] : 0.0f;
#pragma unroll
      for (int v = 0; v < 8; ++v) {
        int m = m0 + mf * 16 + v + 8 * half;
        float x = acc[mf][j][v] + bv;
        if (relu) x = x > 0.0f ? x : 0.0f;
        size_t idx = (size_t)m * ldc + n;
        if (resid) x += resid[idx];
        if (Cf) Cf[idx] = x;
        if (Cb) Cb[idx] = (bf16)x;
      }
    }
}

// ---------------- self-attn scores: S[b,h,q,k] = (Q.Kh^T)/8 ----------------
// grid = (S/128, S/64, B*H), block 256
__global__ void k_scores(const bf16* __restrict__ Q, const bf16* __restrict__ Kp,
                         float* __restrict__ Sc) {
  int lane = threadIdx.x & 31, wave = threadIdx.x >> 5;
  int bh = blockIdx.z, b = bh >> 3, h = bh & 7;
  int q0 = (blockIdx.x * 8 + wave) * 16;
  int n0 = blockIdx.y * 64;
  const bf16* qb = Q + ((size_t)b * SS) * DD + h * DH;
  const bf16* kb = Kp + ((size_t)b * SS) * DD + h * DH;
  v8f acc[4];
#pragma unroll
  for (int j = 0; j < 4; ++j) acc[j] = (v8f)(0.0f);
  for (int kd = 0; kd < DH; kd += 32) {
    v16bf a = load_a_frag(qb + (size_t)q0 * DD + kd, DD, lane);
#pragma unroll
    for (int j = 0; j < 4; ++j) {
      v16bf bfr = load_b_frag(kb + (size_t)(n0 + j * 16) * DD + kd, DD, lane);
      acc[j] = wmma_bf16(a, bfr, acc[j]);
    }
  }
  int half = lane >> 4, col = lane & 15;
#pragma unroll
  for (int j = 0; j < 4; ++j)
#pragma unroll
    for (int v = 0; v < 8; ++v) {
      int q = q0 + v + 8 * half, n = n0 + j * 16 + col;
      Sc[((size_t)bh * SS + q) * SS + n] = acc[j][v] * 0.125f;
    }
}

// ---------------- masked softmax over rows of 512, f32 -> bf16 probs ----------------
__global__ void k_softmax(const float* __restrict__ Sc, const int* __restrict__ mask,
                          bf16* __restrict__ P) {
  int row = blockIdx.x;            // bh*S + q
  int bh = row >> 9, q = row & 511, b = bh >> 3;
  const float* src = Sc + (size_t)row * SS;
  const int* mrow = mask + ((size_t)b * SS + q) * SS;
  __shared__ float red[128];
  float vals[4];
  float mx = -3.0e38f;
#pragma unroll
  for (int i = 0; i < 4; ++i) {
    int c = threadIdx.x + i * 128;
    float x = src[c];
    if (mrow[c] != 0) x = -1.0e18f;
    vals[i] = x;
    mx = fmaxf(mx, x);
  }
  red[threadIdx.x] = mx; __syncthreads();
  for (int s2 = 64; s2 > 0; s2 >>= 1) {
    if (threadIdx.x < s2) red[threadIdx.x] = fmaxf(red[threadIdx.x], red[threadIdx.x + s2]);
    __syncthreads();
  }
  mx = red[0]; __syncthreads();
  float sum = 0.0f;
#pragma unroll
  for (int i = 0; i < 4; ++i) { vals[i] = __expf(vals[i] - mx); sum += vals[i]; }
  red[threadIdx.x] = sum; __syncthreads();
  for (int s2 = 64; s2 > 0; s2 >>= 1) {
    if (threadIdx.x < s2) red[threadIdx.x] += red[threadIdx.x + s2];
    __syncthreads();
  }
  float inv = 1.0f / red[0];
#pragma unroll
  for (int i = 0; i < 4; ++i)
    P[(size_t)row * SS + threadIdx.x + i * 128] = (bf16)(vals[i] * inv);
}

// ---------------- ctx[b,q,h*64+d] = sum_k P[b,h,q,k] * V[b,k,h*64+d] ----------------
__global__ void k_attnv(const bf16* __restrict__ P, const bf16* __restrict__ Vt,
                        bf16* __restrict__ Ctx) {
  int lane = threadIdx.x & 31, wave = threadIdx.x >> 5;
  int bh = blockIdx.y, b = bh >> 3, h = bh & 7;
  int q0 = (blockIdx.x * 8 + wave) * 16;
  const bf16* A = P + (size_t)bh * SS * SS + (size_t)q0 * SS;
  const bf16* Bt = Vt + (size_t)bh * DH * SS;
  v8f acc[4];
#pragma unroll
  for (int j = 0; j < 4; ++j) acc[j] = (v8f)(0.0f);
  for (int k = 0; k < SS; k += 32) {
    v16bf a = load_a_frag(A + k, SS, lane);
#pragma unroll
    for (int j = 0; j < 4; ++j) {
      v16bf bfr = load_b_frag(Bt + (size_t)(j * 16) * SS + k, SS, lane);
      acc[j] = wmma_bf16(a, bfr, acc[j]);
    }
  }
  int half = lane >> 4, col = lane & 15;
#pragma unroll
  for (int j = 0; j < 4; ++j)
#pragma unroll
    for (int v = 0; v < 8; ++v) {
      int q = q0 + v + 8 * half, d = j * 16 + col;
      Ctx[((size_t)b * SS + q) * DD + h * DH + d] = (bf16)acc[j][v];
    }
}

// ---------------- cross-attn out-proj fused with segment max aggregation ----------------
__global__ void k_gemm_agg(const bf16* __restrict__ A, const bf16* __restrict__ Bt,
                           const float* __restrict__ bias,
                           const int* __restrict__ cmap, const int* __restrict__ cmask,
                           unsigned* __restrict__ Agg) {
  int lane = threadIdx.x & 31, wave = threadIdx.x >> 5;
  int m0 = (blockIdx.x * 8 + wave) * 16;
  int n0 = blockIdx.y * 64;
  v8f acc[4];
#pragma unroll
  for (int j = 0; j < 4; ++j) acc[j] = (v8f)(0.0f);
  const bf16* arow = A + (size_t)m0 * DD;
  for (int k = 0; k < DD; k += 32) {
    v16bf a = load_a_frag(arow + k, DD, lane);
#pragma unroll
    for (int j = 0; j < 4; ++j) {
      v16bf b = load_b_frag(Bt + (size_t)(n0 + j * 16) * DD + k, DD, lane);
      acc[j] = wmma_bf16(a, b, acc[j]);
    }
  }
  int half = lane >> 4, col = lane & 15;
  float ninf = __uint_as_float(0xFF800000u);
#pragma unroll
  for (int v = 0; v < 8; ++v) {
    int m = m0 + v + 8 * half;
    int tc = m >> 9, s = m & 511;
    int keep = cmask[((size_t)tc * SS + s) * LLN];   // chain_mask[tc,s,0]
    int b = cmap[tc];
#pragma unroll
    for (int j = 0; j < 4; ++j) {
      int n = n0 + j * 16 + col;
      float x = keep ? (acc[j][v] + bias[n]) : ninf;
      atomicMax(&Agg[((size_t)b * SS + s) * DD + n], enc_f32(x));
    }
  }
}

// ---------------- tiny-L cross attention core (L=12 keys, VALU) ----------------
__global__ void k_crossattn(const bf16* __restrict__ Qb, const bf16* __restrict__ Kc,
                            const bf16* __restrict__ Vc, const int* __restrict__ cmap,
                            const int* __restrict__ cmask, bf16* __restrict__ Out) {
  int t = blockIdx.x * 256 + threadIdx.x;     // over TC*S*H
  if (t >= TCC * SS * HH) return;
  int h = t & 7, rest = t >> 3, s = rest & 511, tc = rest >> 9;
  int bm = cmap[tc];
  const bf16* q = Qb + ((size_t)bm * SS + s) * DD + h * DH;
  const bf16* kb = Kc + (size_t)tc * LLN * DD + h * DH;
  const bf16* vb = Vc + (size_t)tc * LLN * DD + h * DH;
  const int* mrow = cmask + ((size_t)tc * SS + s) * LLN;
  float qv[DH];
#pragma unroll
  for (int d = 0; d < DH; ++d) qv[d] = (float)q[d];
  float sc[LLN];
  float mx = -3.0e38f;
  for (int l = 0; l < LLN; ++l) {
    float x;
    if (mrow[l] != 0) x = -1.0e18f;
    else {
      float a = 0.0f;
      for (int d = 0; d < DH; ++d) a += qv[d] * (float)kb[(size_t)l * DD + d];
      x = a * 0.125f;
    }
    sc[l] = x;
    mx = fmaxf(mx, x);
  }
  float sum = 0.0f;
  for (int l = 0; l < LLN; ++l) { sc[l] = __expf(sc[l] - mx); sum += sc[l]; }
  float inv = 1.0f / sum;
  bf16* o = Out + ((size_t)tc * SS + s) * DD + h * DH;
  for (int d = 0; d < DH; ++d) {
    float a = 0.0f;
    for (int l = 0; l < LLN; ++l) a += sc[l] * (float)vb[(size_t)l * DD + d];
    o[d] = (bf16)(a * inv);
  }
}

// ---------------- LayerNorm row of 512 -> bf16 ----------------
__global__ void k_layernorm(const float* __restrict__ X, const float* __restrict__ g,
                            const float* __restrict__ bta, bf16* __restrict__ Y) {
  int row = blockIdx.x;
  const float* x = X + (size_t)row * DD;
  __shared__ float r1[128], r2[128];
  float v[4], s = 0.0f, ss = 0.0f;
#pragma unroll
  for (int i = 0; i < 4; ++i) {
    v[i] = x[threadIdx.x + i * 128];
    s += v[i]; ss += v[i] * v[i];
  }
  r1[threadIdx.x] = s; r2[threadIdx.x] = ss; __syncthreads();
  for (int s2 = 64; s2 > 0; s2 >>= 1) {
    if (threadIdx.x < s2) { r1[threadIdx.x] += r1[threadIdx.x + s2]; r2[threadIdx.x] += r2[threadIdx.x + s2]; }
    __syncthreads();
  }
  float mu = r1[0] * (1.0f / DD);
  float var = r2[0] * (1.0f / DD) - mu * mu;
  float inv = rsqrtf(var + 1e-6f);
#pragma unroll
  for (int i = 0; i < 4; ++i) {
    int c = threadIdx.x + i * 128;
    Y[(size_t)row * DD + c] = (bf16)((v[i] - mu) * inv * g[c] + bta[c]);
  }
}

// ---------------- small elementwise kernels ----------------
__global__ void k_wt(const float* __restrict__ W, bf16* __restrict__ Wt,
                     int din, int dout) {                    // W[din,dout] -> Wt[dout,din]
  size_t i = (size_t)blockIdx.x * 256 + threadIdx.x;
  if (i >= (size_t)din * dout) return;
  int n = (int)(i % dout), k = (int)(i / dout);
  Wt[(size_t)n * din + k] = (bf16)W[i];
}
__global__ void k_f2b(const float* __restrict__ X, bf16* __restrict__ Y, size_t n) {
  size_t i = (size_t)blockIdx.x * 256 + threadIdx.x;
  if (i < n) Y[i] = (bf16)X[i];
}
__global__ void k_vtrans(const bf16* __restrict__ Vs, bf16* __restrict__ Vt) {
  size_t i = (size_t)blockIdx.x * 256 + threadIdx.x;       // over B*S*D
  if (i >= (size_t)BS * DD) return;
  int d = (int)(i % DD);
  size_t bs = i / DD;
  int s = (int)(bs % SS), b = (int)(bs / SS);
  int h = d >> 6, dd = d & 63;
  Vt[(((size_t)(b * HH + h)) * DH + dd) * SS + s] = Vs[i];
}
__global__ void k_fill_u32(unsigned* __restrict__ P, unsigned val, size_t n) {
  size_t i = (size_t)blockIdx.x * 256 + threadIdx.x;
  if (i < n) P[i] = val;
}
__global__ void k_prepgate(const unsigned* __restrict__ Agg, const float* __restrict__ AttnC,
                           float* __restrict__ CtxF, bf16* __restrict__ CatA) {
  size_t i = (size_t)blockIdx.x * 256 + threadIdx.x;       // over B*S*D
  if (i >= (size_t)BS * DD) return;
  int c = (int)(i % DD);
  size_t row = i / DD;
  float v = dec_f32(Agg[i]);
  if (v + 1.0f == v) v = 0.0f;                              // zero -inf / huge entries
  CtxF[i] = v;
  CatA[row * (2 * DD) + c] = (bf16)AttnC[i];
  CatA[row * (2 * DD) + DD + c] = (bf16)v;
}
__global__ void k_gate(const float* __restrict__ Zpre, const float* __restrict__ AttnC,
                       const float* __restrict__ CtxF, const float* __restrict__ Inp,
                       float* __restrict__ Out1) {
  size_t i = (size_t)blockIdx.x * 256 + threadIdx.x;
  if (i >= (size_t)BS * DD) return;
  float z = 1.0f / (1.0f + __expf(-Zpre[i]));
  Out1[i] = z * AttnC[i] + (1.0f - z) * CtxF[i] + Inp[i];
}

// ---------------- host orchestration ----------------
extern "C" void kernel_launch(void* const* d_in, const int* in_sizes, int n_in,
                              void* d_out, int out_size, void* d_ws, size_t ws_size,
                              hipStream_t stream) {
  (void)in_sizes; (void)n_in; (void)out_size; (void)ws_size;
  const float* inputs    = (const float*)d_in[0];
  const int*   attn_mask = (const int*)d_in[1];
  const int*   chain_map = (const int*)d_in[2];
  const float* span_emb  = (const float*)d_in[3];
  const int*   chain_mask= (const int*)d_in[4];
  const float* ln_g  = (const float*)d_in[5];
  const float* ln_b  = (const float*)d_in[6];
  const float* sa_wq = (const float*)d_in[7];  const float* sa_bq = (const float*)d_in[8];
  const float* sa_wk = (const float*)d_in[9];  const float* sa_bk = (const float*)d_in[10];
  const float* sa_wv = (const float*)d_in[11]; const float* sa_bv = (const float*)d_in[12];
  const float* sa_wo = (const float*)d_in[13]; const float* sa_bo = (const float*)d_in[14];
  const float* lc_w  = (const float*)d_in[15]; const float* lc_b  = (const float*)d_in[16];
  const float* ca_wq = (const float*)d_in[17]; const float* ca_bq = (const float*)d_in[18];
  const float* ca_wk = (const float*)d_in[19]; const float* ca_bk = (const float*)d_in[20];
  const float* ca_wv = (const float*)d_in[21]; const float* ca_bv = (const float*)d_in[22];
  const float* ca_wo = (const float*)d_in[23]; const float* ca_bo = (const float*)d_in[24];
  const float* g_w   = (const float*)d_in[25]; const float* g_b   = (const float*)d_in[26];
  const float* ff_ln_g = (const float*)d_in[27]; const float* ff_ln_b = (const float*)d_in[28];
  const float* ff_w1 = (const float*)d_in[29]; const float* ff_b1 = (const float*)d_in[30];
  const float* ff_w2 = (const float*)d_in[31]; const float* ff_b2 = (const float*)d_in[32];
  float* out = (float*)d_out;

  // --- workspace bump allocator ---
  char* wp = (char*)d_ws;
  auto alloc = [&](size_t bytes) -> char* {
    char* r = wp;
    wp += (bytes + 255) & ~(size_t)255;
    return r;
  };
  bf16* xnorm  = (bf16*)alloc((size_t)BS * DD * 2);
  bf16* wq_t   = (bf16*)alloc((size_t)DD * DD * 2);
  bf16* wk_t   = (bf16*)alloc((size_t)DD * DD * 2);
  bf16* wv_t   = (bf16*)alloc((size_t)DD * DD * 2);
  bf16* wo_t   = (bf16*)alloc((size_t)DD * DD * 2);
  bf16* lc_wt  = (bf16*)alloc((size_t)DD * DCC * 2);
  bf16* cq_t   = (bf16*)alloc((size_t)DD * DD * 2);
  bf16* ck_t   = (bf16*)alloc((size_t)DD * DD * 2);
  bf16* cv_t   = (bf16*)alloc((size_t)DD * DD * 2);
  bf16* co_t   = (bf16*)alloc((size_t)DD * DD * 2);
  bf16* g_wt   = (bf16*)alloc((size_t)DD * 2 * DD * 2);
  bf16* w1_t   = (bf16*)alloc((size_t)DFF2 * DD * 2);
  bf16* w2_t   = (bf16*)alloc((size_t)DD * DFF2 * 2);
  bf16* span_b = (bf16*)alloc((size_t)TCL * DCC * 2);
  bf16* Qs     = (bf16*)alloc((size_t)BS * DD * 2);
  bf16* Ks     = (bf16*)alloc((size_t)BS * DD * 2);
  bf16* Vs     = (bf16*)alloc((size_t)BS * DD * 2);
  bf16* Vt     = (bf16*)alloc((size_t)BS * DD * 2);
  float* Sc    = (float*)alloc((size_t)BB * HH * SS * SS * 4);
  bf16* Pr     = (bf16*)alloc((size_t)BB * HH * SS * SS * 2);
  bf16* ctxb   = (bf16*)alloc((size_t)BS * DD * 2);
  float* attnc = (float*)alloc((size_t)BS * DD * 4);
  bf16* embt   = (bf16*)alloc((size_t)TCL * DD * 2);
  bf16* caK    = (bf16*)alloc((size_t)TCL * DD * 2);
  bf16* caV    = (bf16*)alloc((size_t)TCL * DD * 2);
  bf16* qb     = (bf16*)alloc((size_t)BS * DD * 2);
  bf16* xctx   = (bf16*)alloc((size_t)TCS * DD * 2);
  unsigned* agg= (unsigned*)alloc((size_t)BS * DD * 4);
  float* ctxf  = (float*)alloc((size_t)BS * DD * 4);
  bf16* catA   = (bf16*)alloc((size_t)BS * 2 * DD * 2);
  float* zpre  = (float*)alloc((size_t)BS * DD * 4);
  float* out1  = (float*)alloc((size_t)BS * DD * 4);
  bf16* ybf    = (bf16*)alloc((size_t)BS * DD * 2);
  bf16* inter  = (bf16*)alloc((size_t)BS * DFF2 * 2);

  const dim3 blk256(256), blk128(128);
  auto gceil = [](size_t n) { return (unsigned)((n + 255) / 256); };

  // 1) LayerNorm(inputs) -> bf16
  k_layernorm<<<BS, blk128, 0, stream>>>(inputs, ln_g, ln_b, xnorm);

  // 2) weight transposes to bf16 [dout,din]
  k_wt<<<gceil((size_t)DD * DD), blk256, 0, stream>>>(sa_wq, wq_t, DD, DD);
  k_wt<<<gceil((size_t)DD * DD), blk256, 0, stream>>>(sa_wk, wk_t, DD, DD);
  k_wt<<<gceil((size_t)DD * DD), blk256, 0, stream>>>(sa_wv, wv_t, DD, DD);
  k_wt<<<gceil((size_t)DD * DD), blk256, 0, stream>>>(sa_wo, wo_t, DD, DD);
  k_wt<<<gceil((size_t)DCC * DD), blk256, 0, stream>>>(lc_w, lc_wt, DCC, DD);
  k_wt<<<gceil((size_t)DD * DD), blk256, 0, stream>>>(ca_wq, cq_t, DD, DD);
  k_wt<<<gceil((size_t)DD * DD), blk256, 0, stream>>>(ca_wk, ck_t, DD, DD);
  k_wt<<<gceil((size_t)DD * DD), blk256, 0, stream>>>(ca_wv, cv_t, DD, DD);
  k_wt<<<gceil((size_t)DD * DD), blk256, 0, stream>>>(ca_wo, co_t, DD, DD);
  k_wt<<<gceil((size_t)2 * DD * DD), blk256, 0, stream>>>(g_w, g_wt, 2 * DD, DD);
  k_wt<<<gceil((size_t)DD * DFF2), blk256, 0, stream>>>(ff_w1, w1_t, DD, DFF2);
  k_wt<<<gceil((size_t)DFF2 * DD), blk256, 0, stream>>>(ff_w2, w2_t, DFF2, DD);
  k_f2b<<<gceil((size_t)TCL * DCC), blk256, 0, stream>>>(span_emb, span_b, (size_t)TCL * DCC);

  // 3) self-attn projections (WMMA + TDM-staged B)
  dim3 gP(BS / 256, DD / 64);
  k_gemm<<<gP, blk256, 0, stream>>>(xnorm, DD, wq_t, DD, sa_bq, nullptr, Qs, DD, nullptr, DD, 0);
  k_gemm<<<gP, blk256, 0, stream>>>(xnorm, DD, wk_t, DD, sa_bk, nullptr, Ks, DD, nullptr, DD, 0);
  k_gemm<<<gP, blk256, 0, stream>>>(xnorm, DD, wv_t, DD, sa_bv, nullptr, Vs, DD, nullptr, DD, 0);
  k_vtrans<<<gceil((size_t)BS * DD), blk256, 0, stream>>>(Vs, Vt);

  // 4) scores -> masked softmax -> P@V (WMMA)
  k_scores<<<dim3(SS / 128, SS / 64, BB * HH), blk256, 0, stream>>>(Qs, Ks, Sc);
  k_softmax<<<BB * HH * SS, blk128, 0, stream>>>(Sc, attn_mask, Pr);
  k_attnv<<<dim3(SS / 128, BB * HH), blk256, 0, stream>>>(Pr, Vt, ctxb);
  k_gemm<<<gP, blk256, 0, stream>>>(ctxb, DD, wo_t, DD, sa_bo, attnc, nullptr, DD, nullptr, DD, 0);

  // 5) cross-attn: emb projection, K/V, Q base (WMMA)
  dim3 gE(TCL / 256, DD / 64);
  k_gemm<<<gE, blk256, 0, stream>>>(span_b, DCC, lc_wt, DCC, lc_b, nullptr, embt, DD, nullptr, DCC, 0);
  k_gemm<<<gE, blk256, 0, stream>>>(embt, DD, ck_t, DD, ca_bk, nullptr, caK, DD, nullptr, DD, 0);
  k_gemm<<<gE, blk256, 0, stream>>>(embt, DD, cv_t, DD, ca_bv, nullptr, caV, DD, nullptr, DD, 0);
  k_gemm<<<gP, blk256, 0, stream>>>(xnorm, DD, cq_t, DD, ca_bq, nullptr, qb, DD, nullptr, DD, 0);

  // 6) tiny-L attention core, then fused out-proj + segment-max aggregation
  k_crossattn<<<gceil((size_t)TCC * SS * HH), blk256, 0, stream>>>(qb, caK, caV, chain_map,
                                                                   chain_mask, xctx);
  k_fill_u32<<<gceil((size_t)BS * DD), blk256, 0, stream>>>(agg, 0x007FFFFFu, (size_t)BS * DD);
  k_gemm_agg<<<dim3(TCS / 128, DD / 64), blk256, 0, stream>>>(xctx, co_t, ca_bo, chain_map,
                                                              chain_mask, agg);

  // 7) gate: z = sigmoid([attn,ctx] @ g_w + g_b); out1 = z*attn + (1-z)*ctx + inputs
  k_prepgate<<<gceil((size_t)BS * DD), blk256, 0, stream>>>(agg, attnc, ctxf, catA);
  k_gemm<<<gP, blk256, 0, stream>>>(catA, 2 * DD, g_wt, 2 * DD, g_b, zpre, nullptr, DD,
                                    nullptr, 2 * DD, 0);
  k_gate<<<gceil((size_t)BS * DD), blk256, 0, stream>>>(zpre, attnc, ctxf, inputs, out1);

  // 8) FFN: LN -> relu(y@w1+b1) -> @w2+b2 + out1  (WMMA)
  k_layernorm<<<BS, blk128, 0, stream>>>(out1, ff_ln_g, ff_ln_b, ybf);
  k_gemm<<<dim3(BS / 256, DFF2 / 64), blk256, 0, stream>>>(ybf, DD, w1_t, DD, ff_b1, nullptr,
                                                           inter, DFF2, nullptr, DD, 1);
  k_gemm<<<gP, blk256, 0, stream>>>(inter, DFF2, w2_t, DFF2, ff_b2, out, nullptr, DD, out1,
                                    DFF2, 0);
}
